// Attention_50534585205062
// MI455X (gfx1250) — compile-verified, matrix operand order checked by reading, not run on hardware
//
#include <hip/hip_runtime.h>
#include <math.h>

typedef _Float16 v16h __attribute__((ext_vector_type(16)));
typedef _Float16 v8h  __attribute__((ext_vector_type(8)));
typedef float    v8f  __attribute__((ext_vector_type(8)));

union AFrag { v16h v; v8h p[2]; _Float16 h[16]; };
union CFrag { v8f  v; float f[8]; };
union H8    { v8h  v; _Float16 h[8]; };

#define B_  16
#define C_  128
#define N_  1024
#define HW  32

__device__ __forceinline__ int kbase_of(int i, int kh) {
  return 2 * i + ((i >= 4) ? 8 : 0) + kh * 8;
}

// ---------------------------------------------------------------------------
// Channel GEMM, LDS-staged: Y[b][o][n] = sum_ci W[o][ci] * X[b][ci][n]
// Block = 256 thr (8 waves) owns 64(M) x 32(N); per K-step the block stages
// A(64x32) and B(32x32, transposed) as f16 in LDS; each wave runs one
// v_wmma_f32_16x16x32_f16 tile fed by two ds_load_b128 per operand.
// mode: 0 = none, 1 = +bias, 2 = gelu(acc+bias)*mul, 3 = acc+bias+add
// ---------------------------------------------------------------------------
__global__ void k_gemm_wx(const float* __restrict__ W, const float* __restrict__ X,
                          float* __restrict__ Y, const float* __restrict__ bias,
                          const float* __restrict__ mul, const float* __restrict__ add,
                          int O, int Ci, int mode)
{
  __shared__ _Float16 lA[64 * 32];   // [m][k] row-major
  __shared__ _Float16 lB[32 * 32];   // [n][k] (transposed so lane reads are contiguous)

  int blocksPerB = (O >> 6) * (N_ >> 5);
  int b   = blockIdx.x / blocksPerB;
  int rem = blockIdx.x % blocksPerB;
  int m0blk = (rem / (N_ >> 5)) << 6;
  int n0blk = (rem % (N_ >> 5)) << 5;

  int t    = threadIdx.x;
  int lane = t & 31;
  int wave = t >> 5;
  int mr = lane & 15, kh = lane >> 4;
  int mt = wave >> 1;            // 0..3  (16-row tile inside 64)
  int nt = wave & 1;             // 0..1  (16-col tile inside 32)

  // staging coordinates
  int ar = t >> 2;               // A row 0..63
  int ac = (t & 3) << 3;         // A col {0,8,16,24}
  int bk = t >> 3;               // B k   0..31
  int bn = (t & 7) << 2;         // B n   {0,4,...,28}

  const float* Xb = X + (size_t)b * Ci * N_;
  const float* aSrc = W + (size_t)(m0blk + ar) * Ci + ac;
  const float* bSrc = Xb + (size_t)bk * N_ + n0blk + bn;

  CFrag acc;
  #pragma unroll
  for (int r = 0; r < 8; ++r) acc.f[r] = 0.f;

  for (int k0 = 0; k0 < Ci; k0 += 32) {
    if (k0 + 32 < Ci) {                      // prefetch next K-tile (L2 hot)
      __builtin_prefetch(aSrc + 32, 0, 1);
      __builtin_prefetch(bSrc + (size_t)32 * N_, 0, 1);
    }
    // ---- stage A (64x32 f32 -> f16), coalesced float4 pairs ----
    {
      float4 x0 = *(const float4*)(aSrc);
      float4 x1 = *(const float4*)(aSrc + 4);
      H8 pk;
      pk.h[0] = (_Float16)x0.x; pk.h[1] = (_Float16)x0.y;
      pk.h[2] = (_Float16)x0.z; pk.h[3] = (_Float16)x0.w;
      pk.h[4] = (_Float16)x1.x; pk.h[5] = (_Float16)x1.y;
      pk.h[6] = (_Float16)x1.z; pk.h[7] = (_Float16)x1.w;
      *(v8h*)(&lA[ar * 32 + ac]) = pk.v;
    }
    // ---- stage B (32x32 f32 -> f16), transposed into [n][k] ----
    {
      float4 xv = *(const float4*)(bSrc);
      lB[(bn + 0) * 32 + bk] = (_Float16)xv.x;
      lB[(bn + 1) * 32 + bk] = (_Float16)xv.y;
      lB[(bn + 2) * 32 + bk] = (_Float16)xv.z;
      lB[(bn + 3) * 32 + bk] = (_Float16)xv.w;
    }
    __syncthreads();

    AFrag af, bf;
    af.p[0] = *(const v8h*)(&lA[(mt * 16 + mr) * 32 + kh * 8]);
    af.p[1] = *(const v8h*)(&lA[(mt * 16 + mr) * 32 + 16 + kh * 8]);
    bf.p[0] = *(const v8h*)(&lB[(nt * 16 + mr) * 32 + kh * 8]);
    bf.p[1] = *(const v8h*)(&lB[(nt * 16 + mr) * 32 + 16 + kh * 8]);
    acc.v = __builtin_amdgcn_wmma_f32_16x16x32_f16(false, af.v, false, bf.v,
                                                   (short)0, acc.v, false, false);
    __syncthreads();
    aSrc += 32;
    bSrc += (size_t)32 * N_;
  }

  int col = n0blk + nt * 16 + mr;
  #pragma unroll
  for (int r = 0; r < 8; ++r) {
    int row = m0blk + mt * 16 + r + 8 * kh;
    size_t oidx = ((size_t)b * O + row) * N_ + col;
    float v = acc.f[r];
    if (mode == 1) {
      v += bias[row];
    } else if (mode == 2) {
      v += bias[row];
      v = 0.5f * v * (1.f + erff(v * 0.70710678118654752f));   // exact gelu
      v *= mul[oidx];
    } else if (mode == 3) {
      v += bias[row] + add[oidx];
    }
    Y[oidx] = v;
  }
}

// ---------------------------------------------------------------------------
// Depthwise 3x3 conv, 384 channels, zero pad
// ---------------------------------------------------------------------------
__global__ void k_dwconv(const float* __restrict__ in, const float* __restrict__ w,
                         float* __restrict__ out)
{
  int idx = blockIdx.x * blockDim.x + threadIdx.x;
  if (idx >= B_ * 384 * N_) return;
  int n  = idx & (N_ - 1);
  int ch = (idx >> 10) % 384;
  int b  = idx / (384 * N_);
  int y = n >> 5, x = n & 31;
  const float* ip = in + ((size_t)b * 384 + ch) * N_;
  const float* wp = w + ch * 9;
  float s = 0.f;
  #pragma unroll
  for (int dy = 0; dy < 3; ++dy) {
    int yy = y + dy - 1;
    if (yy < 0 || yy >= HW) continue;
    #pragma unroll
    for (int dx = 0; dx < 3; ++dx) {
      int xx = x + dx - 1;
      if (xx < 0 || xx >= HW) continue;
      s += ip[yy * HW + xx] * wp[dy * 3 + dx];
    }
  }
  out[idx] = s;
}

// ---------------------------------------------------------------------------
// corrcoef/argsort pipeline
// ---------------------------------------------------------------------------
__global__ void k_qmean(const float* __restrict__ qkv2, float* __restrict__ qm)
{
  int idx = blockIdx.x * blockDim.x + threadIdx.x;
  if (idx >= C_ * N_) return;
  int c = idx >> 10, n = idx & (N_ - 1);
  float s = 0.f;
  for (int b = 0; b < B_; ++b) s += qkv2[((size_t)b * 384 + c) * N_ + n];
  qm[idx] = s * (1.f / 16.f);
}

__global__ void k_mu(const float* __restrict__ qm, float* __restrict__ mu)
{
  int c = blockIdx.x * blockDim.x + threadIdx.x;
  if (c >= C_) return;
  float s = 0.f;
  for (int n = 0; n < N_; ++n) s += qm[c * N_ + n];
  mu[c] = s * (1.f / (float)N_);
}

__global__ void k_cov(const float* __restrict__ qm, const float* __restrict__ mu,
                      float* __restrict__ cov)
{
  int idx = blockIdx.x * blockDim.x + threadIdx.x;
  if (idx >= C_ * C_) return;
  int c = idx >> 7, d = idx & 127;
  float mc = mu[c], md = mu[d], s = 0.f;
  for (int n = 0; n < N_; ++n)
    s += (qm[c * N_ + n] - mc) * (qm[d * N_ + n] - md);
  cov[idx] = s;
}

__global__ void k_sd(const float* __restrict__ cov, float* __restrict__ sd)
{
  int c = blockIdx.x * blockDim.x + threadIdx.x;
  if (c >= C_) return;
  sd[c] = sqrtf(cov[c * C_ + c]);
}

__global__ void k_score(const float* __restrict__ cov, const float* __restrict__ sd,
                        float* __restrict__ score)
{
  int c = blockIdx.x * blockDim.x + threadIdx.x;
  if (c >= C_) return;
  float s = 0.f, ic = 1.f / sd[c];
  for (int d = 0; d < C_; ++d) s += cov[c * C_ + d] * ic / sd[d];
  score[c] = s * (1.f / (float)C_);
}

__global__ void k_order(const float* __restrict__ score, int* __restrict__ order)
{
  int c = threadIdx.x;
  if (c >= C_) return;
  float sc = score[c];
  int rank = 0;
  for (int j = 0; j < C_; ++j) {
    float sj = score[j];
    if (sj > sc || (sj == sc && j < c)) ++rank;   // stable descending argsort
  }
  order[rank] = c;
}

__global__ void k_permute(const float* __restrict__ qkv2, const int* __restrict__ order,
                          float* __restrict__ perm)
{
  int idx = blockIdx.x * blockDim.x + threadIdx.x;
  if (idx >= 3 * B_ * C_ * N_) return;
  int which = idx / (B_ * C_ * N_);
  int rem = idx % (B_ * C_ * N_);
  int b = rem >> 17;
  int c = (rem >> 10) & 127;
  int n = rem & (N_ - 1);
  perm[idx] = qkv2[((size_t)b * 384 + which * C_ + order[c]) * N_ + n];
}

// ---------------------------------------------------------------------------
// RPE bias: separable features, full 1024x1024 bias, antialiased resize
// ---------------------------------------------------------------------------
__global__ void k_smean(const float* __restrict__ sm, float* __restrict__ s)
{
  int n = blockIdx.x * blockDim.x + threadIdx.x;
  if (n >= N_) return;
  float acc = 0.f;
  for (int b = 0; b < B_; ++b) acc += sm[b * N_ + n];
  s[n] = acc * (1.f / 16.f);
}

__global__ void k_rpe_ab(const float* __restrict__ W1, const float* __restrict__ b1,
                         const float* __restrict__ me, const int* __restrict__ qmod,
                         const int* __restrict__ kmod, const float* __restrict__ s,
                         float* __restrict__ afeat, float* __restrict__ bfeat)
{
  int idx = blockIdx.x * blockDim.x + threadIdx.x;
  if (idx >= N_ * 64) return;
  int i = idx >> 6, d = idx & 63;
  int ix = i & 31, iy = i >> 5;
  float xx = -0.5f + ix * (1.f / 31.f);
  float yy = -0.5f + iy * (1.f / 31.f);
  float w0 = W1[0 * 64 + d], w1 = W1[1 * 64 + d];
  int qm_ = qmod[0], km_ = kmod[0];
  float mt = 0.f;
  for (int t = 0; t < 16; ++t) mt += me[qm_ * 16 + t] * W1[(2 + t)  * 64 + d];
  for (int t = 0; t < 16; ++t) mt += me[km_ * 16 + t] * W1[(18 + t) * 64 + d];
  float si = s[i];
  afeat[idx] =  xx * w0 + yy * w1 + si * W1[34 * 64 + d];
  bfeat[idx] = -xx * w0 - yy * w1 + si * W1[35 * 64 + d] + mt + b1[d];
}

__global__ void k_bias_full(const float* __restrict__ afeat, const float* __restrict__ bfeat,
                            const float* __restrict__ W2, const float* __restrict__ b2,
                            float* __restrict__ bias)
{
  int idx = blockIdx.x * blockDim.x + threadIdx.x;
  if (idx >= N_ * N_) return;
  int i = idx >> 10, j = idx & (N_ - 1);
  const float* ai = afeat + i * 64;
  const float* bj = bfeat + j * 64;
  float s = 0.f;
  #pragma unroll
  for (int d = 0; d < 64; ++d) s += fmaxf(ai[d] + bj[d], 0.f) * W2[d];
  bias[idx] = s + b2[0];
}

// jax.image.resize(method='bilinear', antialias) -- separable triangle kernel
__global__ void k_resize_rows(const float* __restrict__ src, float* __restrict__ dst, int gc)
{
  int idx = blockIdx.x * blockDim.x + threadIdx.x;
  if (idx >= gc * N_) return;
  int p = idx / N_, j = idx % N_;
  float scale = (float)gc / (float)N_;
  float sample = (p + 0.5f) / scale - 0.5f;
  float radius = 1.f / scale;
  int i0 = (int)ceilf(sample - radius);  if (i0 < 0)      i0 = 0;
  int i1 = (int)floorf(sample + radius); if (i1 > N_ - 1) i1 = N_ - 1;
  float acc = 0.f, wsum = 0.f;
  for (int i = i0; i <= i1; ++i) {
    float wgt = 1.f - fabsf((float)i - sample) * scale;
    if (wgt > 0.f) { acc += wgt * src[i * N_ + j]; wsum += wgt; }
  }
  dst[p * N_ + j] = acc / wsum;
}

__global__ void k_resize_cols(const float* __restrict__ src, float* __restrict__ dst, int gc)
{
  int idx = blockIdx.x * blockDim.x + threadIdx.x;
  if (idx >= gc * gc) return;
  int p = idx / gc, q = idx % gc;
  float scale = (float)gc / (float)N_;
  float sample = (q + 0.5f) / scale - 0.5f;
  float radius = 1.f / scale;
  int j0 = (int)ceilf(sample - radius);  if (j0 < 0)      j0 = 0;
  int j1 = (int)floorf(sample + radius); if (j1 > N_ - 1) j1 = N_ - 1;
  float acc = 0.f, wsum = 0.f;
  for (int j = j0; j <= j1; ++j) {
    float wgt = 1.f - fabsf((float)j - sample) * scale;
    if (wgt > 0.f) { acc += wgt * src[p * N_ + j]; wsum += wgt; }
  }
  dst[p * gc + q] = acc / wsum;
}

// ---------------------------------------------------------------------------
// Grouped channel attention
// ---------------------------------------------------------------------------
__global__ void k_norms(const float* __restrict__ qp, const float* __restrict__ kp,
                        float* __restrict__ rnq, float* __restrict__ rnk)
{
  int idx = blockIdx.x * blockDim.x + threadIdx.x;
  if (idx >= 2 * B_ * C_) return;
  int which = idx >> 11;
  int t = idx & 2047;
  const float* base = (which ? kp : qp) + (size_t)t * N_;
  float s = 0.f;
  for (int n = 0; n < N_; ++n) s += base[n] * base[n];
  float rn = 1.f / fmaxf(sqrtf(s), 1e-12f);
  if (which) rnk[t] = rn; else rnq[t] = rn;
}

// logits[b][cq][ck] = temp[g] * sum_n (qp*rnq)(kp*rnk)  (WMMA, K = 1024)
__global__ void k_attn_qk(const float* __restrict__ qp, const float* __restrict__ kp,
                          const float* __restrict__ rnq, const float* __restrict__ rnk,
                          const float* __restrict__ temp, float* __restrict__ attn)
{
  const int s0a[4]  = {0, 16, 48, 80};
  const int gca[4]  = {16, 32, 32, 48};
  const int goff[4] = {0, 4096, 20480, 36864};
  const int cum[5]  = {0, 16, 80, 144, 288};
  int lane = threadIdx.x & 31;
  int wave = threadIdx.x >> 5;
  int tile = blockIdx.x * (blockDim.x >> 5) + wave;
  if (tile >= 288) return;
  int g = 0;
  while (tile >= cum[g + 1]) ++g;
  int local = tile - cum[g];
  int gc = gca[g], s0 = s0a[g];
  int gt = gc >> 4, tpb = gt * gt;
  int b  = local / tpb;
  int tt = local % tpb;
  int m0 = (tt / gt) << 4;
  int n0 = (tt % gt) << 4;
  int mr = lane & 15, kh = lane >> 4;
  int rowq = s0 + m0 + mr, rowk = s0 + n0 + mr;
  const float* qrow = qp + ((size_t)b * C_ + rowq) * N_;
  const float* krow = kp + ((size_t)b * C_ + rowk) * N_;
  float sq = rnq[b * C_ + rowq];
  float sk = rnk[b * C_ + rowk];

  CFrag acc;
  #pragma unroll
  for (int r = 0; r < 8; ++r) acc.f[r] = 0.f;
  for (int k0 = 0; k0 < N_; k0 += 32) {
    AFrag af, bf;
    #pragma unroll
    for (int i = 0; i < 8; ++i) {
      int kb = k0 + kbase_of(i, kh);
      af.h[2*i]   = (_Float16)(qrow[kb]     * sq);
      af.h[2*i+1] = (_Float16)(qrow[kb + 1] * sq);
      bf.h[2*i]   = (_Float16)(krow[kb]     * sk);
      bf.h[2*i+1] = (_Float16)(krow[kb + 1] * sk);
    }
    acc.v = __builtin_amdgcn_wmma_f32_16x16x32_f16(false, af.v, false, bf.v,
                                                   (short)0, acc.v, false, false);
  }
  float tg = temp[g];
  #pragma unroll
  for (int r = 0; r < 8; ++r) {
    int row = m0 + r + 8 * kh;
    attn[goff[g] + b * gc * gc + row * gc + n0 + mr] = acc.f[r] * tg;
  }
}

__global__ void k_softmax(float* __restrict__ attn, const float* __restrict__ bg16,
                          const float* __restrict__ bg32, const float* __restrict__ bg48)
{
  const int gca[4]  = {16, 32, 32, 48};
  const int goff[4] = {0, 4096, 20480, 36864};
  const int rcum[5] = {0, 256, 768, 1280, 2048};
  int t = blockIdx.x * blockDim.x + threadIdx.x;
  if (t >= 2048) return;
  int g = 0;
  while (t >= rcum[g + 1]) ++g;
  int local = t - rcum[g];
  int gc = gca[g];
  int b = local / gc, row = local % gc;
  float* p = attn + goff[g] + b * gc * gc + row * gc;
  const float* bg = (g == 0) ? bg16 : ((g == 3) ? bg48 : bg32);
  const float* br = bg + row * gc;
  float buf[48];
  float mx = -3.4e38f;
  for (int j = 0; j < gc; ++j) { float v = p[j] + br[j]; buf[j] = v; if (v > mx) mx = v; }
  float s = 0.f;
  for (int j = 0; j < gc; ++j) { float e = expf(buf[j] - mx); buf[j] = e; s += e; }
  float inv = 1.f / s;
  for (int j = 0; j < gc; ++j) p[j] = buf[j] * inv;
}

// out[b][s0+cq][n] = sum_ck attn * v   (WMMA, K = gc zero-padded to 32)
__global__ void k_attn_v(const float* __restrict__ attn, const float* __restrict__ vp,
                         float* __restrict__ out)
{
  const int s0a[4]  = {0, 16, 48, 80};
  const int gca[4]  = {16, 32, 32, 48};
  const int goff[4] = {0, 4096, 20480, 36864};
  const int cum[5]  = {0, 1024, 3072, 5120, 8192};
  int lane = threadIdx.x & 31;
  int wave = threadIdx.x >> 5;
  int tile = blockIdx.x * (blockDim.x >> 5) + wave;
  if (tile >= 8192) return;
  int g = 0;
  while (tile >= cum[g + 1]) ++g;
  int local = tile - cum[g];
  int gc = gca[g], s0 = s0a[g];
  int gt = gc >> 4, tpb = gt * 64;
  int b  = local / tpb;
  int tt = local % tpb;
  int m0 = (tt / 64) << 4;
  int n0 = (tt % 64) << 4;
  int mr = lane & 15, kh = lane >> 4;
  const float* arow = attn + goff[g] + (size_t)b * gc * gc + (m0 + mr) * gc;
  const float* vb   = vp + ((size_t)b * C_ + s0) * N_;

  CFrag acc;
  #pragma unroll
  for (int r = 0; r < 8; ++r) acc.f[r] = 0.f;
  for (int k0 = 0; k0 < gc; k0 += 32) {
    AFrag af, bf;
    #pragma unroll
    for (int i = 0; i < 8; ++i) {
      int kb = k0 + kbase_of(i, kh);
      af.h[2*i]   = (kb     < gc) ? (_Float16)arow[kb]     : (_Float16)0.f;
      af.h[2*i+1] = (kb + 1 < gc) ? (_Float16)arow[kb + 1] : (_Float16)0.f;
      bf.h[2*i]   = (kb     < gc) ? (_Float16)vb[(size_t)kb       * N_ + n0 + mr] : (_Float16)0.f;
      bf.h[2*i+1] = (kb + 1 < gc) ? (_Float16)vb[(size_t)(kb + 1) * N_ + n0 + mr] : (_Float16)0.f;
    }
    acc.v = __builtin_amdgcn_wmma_f32_16x16x32_f16(false, af.v, false, bf.v,
                                                   (short)0, acc.v, false, false);
  }
  #pragma unroll
  for (int r = 0; r < 8; ++r) {
    int row = s0 + m0 + r + 8 * kh;
    out[((size_t)b * C_ + row) * N_ + n0 + mr] = acc.f[r];
  }
}

__global__ void k_xs(const float* __restrict__ out, const float* __restrict__ qp,
                     const float* __restrict__ kp, const float* __restrict__ rnq,
                     const float* __restrict__ rnk, float* __restrict__ xs)
{
  int idx = blockIdx.x * blockDim.x + threadIdx.x;
  if (idx >= B_ * C_ * N_) return;
  int bc = idx >> 10;
  xs[idx] = out[idx] + qp[idx] * rnq[bc] + kp[idx] * rnk[bc];
}

// ---------------------------------------------------------------------------
extern "C" void kernel_launch(void* const* d_in, const int* in_sizes, int n_in,
                              void* d_out, int out_size, void* d_ws, size_t ws_size,
                              hipStream_t stream)
{
  (void)in_sizes; (void)n_in; (void)out_size; (void)ws_size;
  const float* x     = (const float*)d_in[0];
  const float* smap  = (const float*)d_in[1];
  const float* Wqkv  = (const float*)d_in[2];
  const float* Wdw   = (const float*)d_in[3];
  const float* Wproj = (const float*)d_in[4];
  const float* temp  = (const float*)d_in[5];
  const float* me    = (const float*)d_in[6];
  const float* W1    = (const float*)d_in[7];
  const float* b1    = (const float*)d_in[8];
  const float* W2    = (const float*)d_in[9];
  const float* b2    = (const float*)d_in[10];
  const float* Wg    = (const float*)d_in[11];
  const float* bg    = (const float*)d_in[12];
  const float* Wd    = (const float*)d_in[13];
  const float* bd    = (const float*)d_in[14];
  const float* Wu    = (const float*)d_in[15];
  const float* bu    = (const float*)d_in[16];
  const int*   qmod  = (const int*)d_in[17];
  const int*   kmod  = (const int*)d_in[18];
  float* yout = (float*)d_out;

  const size_t BCN = (size_t)B_ * C_ * N_;          // 2,097,152
  float* w = (float*)d_ws;
  float* qkv1 = w;                                  // 384*B*N = 6,291,456
  float* qkv2 = qkv1 + 3 * BCN;                     // 6,291,456
  // permuted q/k/v overlay qkv1 (written after qkv1's last read by k_dwconv)
  float* qp = qkv1;
  float* kp = qkv1 + BCN;
  float* vp = qkv1 + 2 * BCN;
  // xs / gated / down overlay qkv2 (dead after k_qmean + k_permute)
  float* xs    = qkv2;
  float* gated = qkv2 + BCN;
  float* down  = qkv2 + 2 * BCN;
  float* S = qkv2 + 3 * BCN;
  float* qm    = S;              S += (size_t)C_ * N_;   // 131072
  float* mu    = S;              S += C_;
  float* cov   = S;              S += C_ * C_;
  float* sd    = S;              S += C_;
  float* score = S;              S += C_;
  float* smean = S;              S += N_;
  int*   order = (int*)S;        S += C_;
  float* afeat = S;              S += (size_t)N_ * 64;
  float* bfeat = S;              S += (size_t)N_ * 64;
  float* biasF = S;              S += (size_t)N_ * N_;   // 1,048,576
  float* rtmp  = S;              S += 48 * N_;
  float* bg16v = S;              S += 16 * 16;
  float* bg32v = S;              S += 32 * 32;
  float* bg48v = S;              S += 48 * 48;
  float* rnq   = S;              S += B_ * C_;
  float* rnk   = S;              S += B_ * C_;
  float* attn  = S;              S += 73728;
  float* outb  = S;              S += BCN;               // attention output / y

  // 1. QKV projection (LDS-staged WMMA GEMM 384x128 @ 128x16384)
  k_gemm_wx<<<B_ * (384 / 64) * (N_ / 32), 256, 0, stream>>>(
      Wqkv, x, qkv1, nullptr, nullptr, nullptr, 384, 128, 0);
  // 2. depthwise 3x3
  k_dwconv<<<(B_ * 384 * N_ + 255) / 256, 256, 0, stream>>>(qkv1, Wdw, qkv2);
  // 3-8. corrcoef -> order
  k_qmean<<<(C_ * N_ + 255) / 256, 256, 0, stream>>>(qkv2, qm);
  k_mu<<<1, 128, 0, stream>>>(qm, mu);
  k_cov<<<(C_ * C_ + 255) / 256, 256, 0, stream>>>(qm, mu, cov);
  k_sd<<<1, 128, 0, stream>>>(cov, sd);
  k_score<<<1, 128, 0, stream>>>(cov, sd, score);
  k_order<<<1, 128, 0, stream>>>(score, order);
  // 9. channel permutation -> qp/kp/vp
  k_permute<<<(3 * (int)BCN + 255) / 256, 256, 0, stream>>>(qkv2, order, qkv1);
  // 10-13. RPE bias + antialiased resizes
  k_smean<<<(N_ + 255) / 256, 256, 0, stream>>>(smap, smean);
  k_rpe_ab<<<(N_ * 64 + 255) / 256, 256, 0, stream>>>(W1, b1, me, qmod, kmod,
                                                      smean, afeat, bfeat);
  k_bias_full<<<(N_ * N_ + 255) / 256, 256, 0, stream>>>(afeat, bfeat, W2, b2, biasF);
  {
    int gcs[3] = {16, 32, 48};
    float* dsts[3] = {bg16v, bg32v, bg48v};
    for (int i = 0; i < 3; ++i) {
      int gc = gcs[i];
      k_resize_rows<<<(gc * N_ + 255) / 256, 256, 0, stream>>>(biasF, rtmp, gc);
      k_resize_cols<<<(gc * gc + 255) / 256, 256, 0, stream>>>(rtmp, dsts[i], gc);
    }
  }
  // 14-17. grouped channel attention (WMMA QK^T and attn*V)
  k_norms<<<(2 * B_ * C_ + 255) / 256, 256, 0, stream>>>(qp, kp, rnq, rnk);
  k_attn_qk<<<(288 + 3) / 4, 128, 0, stream>>>(qp, kp, rnq, rnk, temp, attn);
  k_softmax<<<(2048 + 255) / 256, 256, 0, stream>>>(attn, bg16v, bg32v, bg48v);
  k_attn_v<<<(8192 + 3) / 4, 128, 0, stream>>>(attn, vp, outb);
  // 18. xs = out + qg + kg
  k_xs<<<((int)BCN + 255) / 256, 256, 0, stream>>>(outb, qp, kp, rnq, rnk, xs);
  // 19-22. gated MLP chain + final projection (LDS-staged WMMA, fused epilogues)
  {
    int g128 = B_ * (128 / 64) * (N_ / 32);
    int g64  = B_ * (64 / 64) * (N_ / 32);
    k_gemm_wx<<<g128, 256, 0, stream>>>(Wg, xs, gated, bg, xs, nullptr, 128, 128, 2);
    k_gemm_wx<<<g64, 256, 0, stream>>>(Wd, gated, down, bd, nullptr, nullptr, 64, 128, 1);
    // y = out + (Wu@down + bu), written in place over outb (thread-local RMW)
    k_gemm_wx<<<g128, 256, 0, stream>>>(Wu, down, outb, bu, nullptr, outb, 128, 64, 3);
    k_gemm_wx<<<g128, 256, 0, stream>>>(Wproj, outb, yout, nullptr, nullptr, nullptr,
                                        128, 128, 0);
  }
}
